// NPInstantaneousTransitionPrior_39075612459512
// MI455X (gfx1250) — matrix operand order repaired and checked by reading, not compiled, not run on hardware
//
#include <hip/hip_runtime.h>
#include <math.h>

// Problem constants (from reference)
#define BATCH   32
#define SEQ_T   500
#define DIM     8
#define NWIN    498                 // SEQ_T - L (L=2)
#define NSAMP   (BATCH * NWIN)      // 15936
#define HID     64
#define TAU_F   0.3f
#define ALPHA_F 0.2f

// Packed f16 weight image per MLP (halves): W1a(2048) W1b(2048) W2r(4096)
// W2t(4096) W3r(4096) W3t(4096) W4(64) = 20544 halves = 41,088 B (16B multiple).
#define PACK_H      20544
#define PACK_CHUNKS (PACK_H * 2 / 16)   // 2568 x 16-byte async transfers

typedef __attribute__((ext_vector_type(16))) _Float16 v16h;
typedef __attribute__((ext_vector_type(8)))  _Float16 v8h;
typedef __attribute__((ext_vector_type(8)))  float    v8f;

#define WMMA_F16(A, B, C) \
  __builtin_amdgcn_wmma_f32_16x16x32_f16(false, (A), false, (B), (short)0, (C), false, false)

struct WPtrs { const float* p[64]; };  // 8 MLPs x 4 layers x (W, b)

// --- WMMA fragment index maps (wave32, CDNA5 ISA 7.12.2) ---
// A (16x32 f16): M = lane&15, K = (e&7) + (e>=8?16:0) + (lane>=16?8:0)
//   -> per lane: two contiguous 8-half runs at K=koff and K=16+koff, koff=(lane&16)>>1
// B (32x16 f16): N = lane&15, K as for A  -> two 8-half runs at K=koff and K=16+koff
// C/D (16x16 f32): N = lane&15, M = r + (lane>=16?8:0)

__device__ __forceinline__ v16h frag_cat(v8h a, v8h b) {
  return __builtin_shufflevector(a, b, 0,1,2,3,4,5,6,7,8,9,10,11,12,13,14,15);
}
// two 8-half runs separated by 16 halves (B-fragment from a K-innermost weight tile)
__device__ __forceinline__ v16h load_frag_c16(const _Float16* p) {
  return frag_cat(*(const v8h*)p, *(const v8h*)(p + 16));
}
// two 8-half runs separated by 16 halves (A-fragment from row-major 16x64 tile)
__device__ __forceinline__ v16h load_frag_a(const _Float16* p) {
  return frag_cat(*(const v8h*)p, *(const v8h*)(p + 16));
}

// dst = leakyrelu(src(16x64) @ W(64x64) + bias). Wt is TRANSPOSED (K-innermost).
// Returns 32-bit sign mask, bit (n*8+r) = (preact > 0) at this lane's C slot.
__device__ __forceinline__ unsigned forward_hidden(
    const _Float16* __restrict__ src, const _Float16* __restrict__ Wt,
    const float* __restrict__ bias, _Float16* __restrict__ dst, int lane)
{
  const int m = lane & 15, koff = (lane & 16) >> 1, kb0 = (lane & 16) >> 1;
  v16h aLo = load_frag_a(src + m * 64 + koff);
  v16h aHi = load_frag_a(src + m * 64 + 32 + koff);
  unsigned mask = 0;
#pragma unroll
  for (int n = 0; n < 4; ++n) {
    const int col = n * 16 + m;
    v16h bLo = load_frag_c16(Wt + col * 64 + kb0);
    v16h bHi = load_frag_c16(Wt + col * 64 + 32 + kb0);
    v8f c = {};
    c = WMMA_F16(aLo, bLo, c);
    c = WMMA_F16(aHi, bHi, c);
#pragma unroll
    for (int r = 0; r < 8; ++r) {
      int row = r + koff;
      float pre = c[r] + bias[col];
      bool pos = pre > 0.0f;
      mask |= (pos ? 1u : 0u) << (n * 8 + r);
      dst[row * 64 + col] = (_Float16)(pos ? pre : ALPHA_F * pre);
    }
  }
  return mask;
}

// gdst = (g(16x64) @ W^T) .* deriv(dmask). Wr is ROW-MAJOR (W^T[k][col]=W[col][k],
// K-innermost), so B fragments are contiguous too.
__device__ __forceinline__ void backward_step(
    v16h aLo, v16h aHi, const _Float16* __restrict__ Wr,
    unsigned dmask, _Float16* __restrict__ gdst, int lane)
{
  const int m = lane & 15, koff = (lane & 16) >> 1, kb0 = (lane & 16) >> 1;
#pragma unroll
  for (int n = 0; n < 4; ++n) {
    const int col = n * 16 + m;
    v16h bLo = load_frag_c16(Wr + col * 64 + kb0);
    v16h bHi = load_frag_c16(Wr + col * 64 + 32 + kb0);
    v8f c = {};
    c = WMMA_F16(aLo, bLo, c);
    c = WMMA_F16(aHi, bHi, c);
#pragma unroll
    for (int r = 0; r < 8; ++r) {
      int row = r + koff;
      float dfac = ((dmask >> (n * 8 + r)) & 1u) ? 1.0f : ALPHA_F;
      gdst[row * 64 + col] = (_Float16)(c[r] * dfac);
    }
  }
}

// ---- Prep 1: cmask = cumprod(sigmoid((u+noise)/tau) * (1-I), axis=0) ----
__global__ void cmask_prep(const float* __restrict__ u, const float* __restrict__ noise,
                           float* __restrict__ cmask) {
  __shared__ float msk[64];
  int t = threadIdx.x;
  if (t < 64) {
    int r = t >> 3, c = t & 7;
    float z  = (u[t] + noise[t]) / TAU_F;
    float sg = 1.0f / (1.0f + expf(-z));
    msk[t] = (r == c) ? 0.0f : sg;
  }
  __syncthreads();
  if (t < 64) {
    int r = t >> 3, c = t & 7;
    float pz = 1.0f;
    for (int rr = 0; rr <= r; ++rr) pz *= msk[rr * 8 + c];
    cmask[t] = pz;
  }
}

// ---- Prep 2: pack all weights as f16 in the exact LDS image layout ----
__global__ __launch_bounds__(256) void weights_prep(WPtrs wp, _Float16* __restrict__ pack) {
  const int i      = blockIdx.x;     // 0..7
  const int in_dim = 17 + i;
  const float* W1 = wp.p[i * 8 + 0];
  const float* W2 = wp.p[i * 8 + 2];
  const float* W3 = wp.p[i * 8 + 4];
  const float* W4 = wp.p[i * 8 + 6];
  _Float16* P = pack + (size_t)i * PACK_H;

  for (int t = threadIdx.x; t < 2048; t += 256) {        // W1a [k][n], K padded
    int k = t >> 6, n = t & 63;
    P[t] = (_Float16)(k < in_dim ? W1[k * 64 + n] : 0.0f);
  }
  for (int t = threadIdx.x; t < 2048; t += 256) {        // W1b [n][k]
    int n = t >> 5, k = t & 31;
    P[2048 + t] = (_Float16)(k < in_dim ? W1[k * 64 + n] : 0.0f);
  }
  for (int t = threadIdx.x; t < 4096; t += 256) {
    int n = t >> 6, k = t & 63;
    P[4096  + t] = (_Float16)W2[t];                      // W2r [k][n]
    P[8192  + t] = (_Float16)W2[k * 64 + n];             // W2t [n][k]
    P[12288 + t] = (_Float16)W3[t];                      // W3r
    P[16384 + t] = (_Float16)W3[k * 64 + n];             // W3t
  }
  if (threadIdx.x < 64) P[20480 + threadIdx.x] = (_Float16)W4[threadIdx.x];
}

// ---- Main fused forward + Jacobian kernel ----
// grid = (996/4, 8); block = 128 (4 wave32s). Each wave: one 16-sample tile, MLP i.
__global__ __launch_bounds__(128) void gs_mlp_fused(
    const float* __restrict__ x, WPtrs wp, const float* __restrict__ cmask,
    const _Float16* __restrict__ pack,
    float* __restrict__ slad_part, float* __restrict__ out)
{
  const int i      = blockIdx.y;          // MLP index 0..7
  const int in_dim = 17 + i;
  const int tid    = threadIdx.x;
  const int wave   = tid >> 5;
  const int lane   = tid & 31;
  const int tile   = blockIdx.x * 4 + wave;  // 0..995 (exact)
  const int m      = lane & 15;
  const int koff   = (lane & 16) >> 1;
  const int kb0    = (lane & 16) >> 1;

  // f16 weight image (async-DMA'd from the prepacked global copy) + staging.
  __shared__ __align__(16) _Float16 wsm[PACK_H];   // 41,088 B
  __shared__ float b1s[64], b2s[64], b3s[64];
  __shared__ float b4s;
  __shared__ float cm[8];
  __shared__ __align__(16) _Float16 stA[4][16 * 64];
  __shared__ __align__(16) _Float16 stB[4][16 * 64];

  const _Float16* W1a = wsm;               // row-major [k_in(pad32)][n_hid]
  const _Float16* W1b = wsm + 2048;        // transposed [n_hid][k_in(pad32)]
  const _Float16* W2r = wsm + 4096;
  const _Float16* W2t = wsm + 8192;
  const _Float16* W3r = wsm + 12288;
  const _Float16* W3t = wsm + 16384;
  const _Float16* W4h = wsm + 20480;

  // ---- Async memory->LDS weight DMA (no VGPR round trip, ASYNCcnt tracked) ----
  {
    const _Float16* wsrc = pack + (size_t)i * PACK_H;
    const unsigned lds_base = (unsigned)(uintptr_t)(&wsm[0]);
    for (int t = tid; t < PACK_CHUNKS; t += 128) {
      unsigned boff = (unsigned)t * 16u;
      unsigned laddr = lds_base + boff;
      asm volatile("global_load_async_to_lds_b128 %0, %1, %2"
                   :: "v"(laddr), "v"(boff), "s"(wsrc) : "memory");
    }
  }

  // Small f32 params via the normal path (overlaps with the async DMA).
  const float* B1v = wp.p[i * 8 + 1];
  const float* B2v = wp.p[i * 8 + 3];
  const float* B3v = wp.p[i * 8 + 5];
  const float* B4v = wp.p[i * 8 + 7];
  if (tid < 64) { b1s[tid] = B1v[tid]; b2s[tid] = B2v[tid]; b3s[tid] = B3v[tid]; }
  if (tid == 0) b4s = B4v[0];
  if (tid < 8)  cm[tid] = cmask[i * 8 + tid];

  // ---- Prefetch this lane's input window while the DMA is in flight ----
  // inp[k] = x[b,w,k] (k<8) | x[b,w+1,k-8]*cmask[i,k-8] (k<16) | x[b,w+2,k-16].
  // Window rows are contiguous in x, so inp[k] == xp[k] with per-range scaling.
  const int s0 = tile * 16 + m;
  const int bb = s0 / NWIN, ww = s0 % NWIN;
  const float* xp = x + (bb * SEQ_T + ww) * DIM;
  float xv[16];
#pragma unroll
  for (int e = 0; e < 16; ++e) {
    int k = (e & 7) + ((e & 8) ? 16 : 0) + koff;
    xv[e] = (k < in_dim) ? xp[k] : 0.0f;
  }

  asm volatile("s_wait_asynccnt 0x0" ::: "memory");
  __syncthreads();

  _Float16* sA = stA[wave];
  _Float16* sB = stB[wave];

  // ---- Layer 1: inp(16 x in_dim, K-padded to 32) @ W1 ----
  v16h a0;
#pragma unroll
  for (int e = 0; e < 16; ++e) {
    int k = (e & 7) + ((e & 8) ? 16 : 0) + koff;
    float v = xv[e];
    if (k >= 8 && k < 16) v *= cm[k - 8];
    a0[e] = (_Float16)v;
  }
  unsigned m1 = 0;
#pragma unroll
  for (int n = 0; n < 4; ++n) {
    const int col = n * 16 + m;
    v16h bfr = load_frag_c16(W1b + col * 32 + kb0);   // K in 0..31, contiguous
    v8f c = {};
    c = WMMA_F16(a0, bfr, c);
#pragma unroll
    for (int r = 0; r < 8; ++r) {
      int row = r + koff;
      float pre = c[r] + b1s[col];
      bool pos = pre > 0.0f;
      m1 |= (pos ? 1u : 0u) << (n * 8 + r);
      sA[row * 64 + col] = (_Float16)(pos ? pre : ALPHA_F * pre);
    }
  }
  __builtin_amdgcn_wave_barrier();

  // ---- Layers 2,3 (derivative sign masks stay in VGPRs) ----
  unsigned m2 = forward_hidden(sA, W2t, b2s, sB, lane);
  __builtin_amdgcn_wave_barrier();
  unsigned m3d = forward_hidden(sB, W3t, b3s, sA, lane);  (void)m3d;
  __builtin_amdgcn_wave_barrier();

  // ---- Layer 4: residual = act3 @ W4 + b4 (sA holds act3) ----
  if (lane < 16) {
    float acc = b4s;
#pragma unroll
    for (int kv = 0; kv < 64; kv += 8) {
      v8h av = *(const v8h*)(sA + lane * 64 + kv);
      v8h wv = *(const v8h*)(W4h + kv);
#pragma unroll
      for (int j = 0; j < 8; ++j) acc += (float)av[j] * (float)wv[j];
    }
    out[(tile * 16 + lane) * DIM + i] = acc;
  }

  // ---- Backward: g3 = W4 .* d3 in A layout.  d3 sign == act3 sign (leaky relu). ----
  v16h actLo = load_frag_a(sA + m * 64 + koff);
  v16h actHi = load_frag_a(sA + m * 64 + 32 + koff);
  v16h w4Lo  = load_frag_a(W4h + koff);
  v16h w4Hi  = load_frag_a(W4h + 32 + koff);
  v16h gLo, gHi;
#pragma unroll
  for (int e = 0; e < 16; ++e) {
    gLo[e] = (_Float16)(((float)actLo[e] > 0.0f ? 1.0f : ALPHA_F) * (float)w4Lo[e]);
    gHi[e] = (_Float16)(((float)actHi[e] > 0.0f ? 1.0f : ALPHA_F) * (float)w4Hi[e]);
  }
  backward_step(gLo, gHi, W3r, m2, sB, lane);           // g2 -> sB
  __builtin_amdgcn_wave_barrier();
  gLo = load_frag_a(sB + m * 64 + koff);
  gHi = load_frag_a(sB + m * 64 + 32 + koff);
  backward_step(gLo, gHi, W2r, m1, sA, lane);           // g1 -> sA
  __builtin_amdgcn_wave_barrier();
  gLo = load_frag_a(sA + m * 64 + koff);
  gHi = load_frag_a(sA + m * 64 + 32 + koff);

  // ---- J = g1 @ W1^T : 16 x in_dim (2 N-tiles cover padded 32 cols) ----
  // W1^T[k_hid][col_in] = W1[col_in][k_hid] -> W1a row-major is K-contiguous.
  const int base_i = NSAMP * DIM + 32 + NSAMP * (16 * i + (i * (i - 1)) / 2);
#pragma unroll
  for (int n = 0; n < 2; ++n) {
    const int col = n * 16 + m;
    v16h bLo = load_frag_c16(W1a + col * 64 + kb0);
    v16h bHi = load_frag_c16(W1a + col * 64 + 32 + kb0);
    v8f c = {};
    c = WMMA_F16(gLo, bLo, c);
    c = WMMA_F16(gHi, bHi, c);
#pragma unroll
    for (int r = 0; r < 8; ++r) {
      int row = r + koff;
      int s   = tile * 16 + row;
      if (col == in_dim - 1)
        slad_part[i * NSAMP + s] = logf(fabsf(c[r]));
      else if (col < in_dim - 1)
        out[base_i + s * (in_dim - 1) + col] = c[r];
    }
  }
}

// ---- Deterministic slad reduction: slad[b] = sum over (i, w) of parts ----
__global__ __launch_bounds__(256) void slad_reduce(const float* __restrict__ part,
                                                   float* __restrict__ slad_out) {
  __shared__ float red[256];
  int b = blockIdx.x;
  float acc = 0.0f;
  for (int t = threadIdx.x; t < NWIN * DIM; t += 256) {
    int i = t / NWIN, w = t % NWIN;
    acc += part[i * NSAMP + b * NWIN + w];
  }
  red[threadIdx.x] = acc;
  __syncthreads();
  for (int off = 128; off > 0; off >>= 1) {
    if (threadIdx.x < off) red[threadIdx.x] += red[threadIdx.x + off];
    __syncthreads();
  }
  if (threadIdx.x == 0) slad_out[b] = red[0];
}

extern "C" void kernel_launch(void* const* d_in, const int* in_sizes, int n_in,
                              void* d_out, int out_size, void* d_ws, size_t ws_size,
                              hipStream_t stream) {
  (void)in_sizes; (void)out_size; (void)ws_size;
  const float* x     = (const float*)d_in[0];
  const float* u     = (const float*)d_in[1];
  const float* noise = (const float*)d_in[n_in - 1];   // logistic_noise (last)
  WPtrs wp;
  for (int j = 0; j < 64; ++j) wp.p[j] = (const float*)d_in[2 + j];

  // Workspace layout: cmask(64 f32) | slad parts(8*NSAMP f32) | f16 weight packs
  float* cmask_ws  = (float*)d_ws;
  float* slad_part = cmask_ws + 64;
  _Float16* pack   = (_Float16*)(slad_part + 8 * NSAMP);   // 16B-aligned offset
  float* out = (float*)d_out;

  cmask_prep<<<1, 64, 0, stream>>>(u, noise, cmask_ws);
  weights_prep<<<DIM, 256, 0, stream>>>(wp, pack);
  gs_mlp_fused<<<dim3(NSAMP / 16 / 4, DIM), 128, 0, stream>>>(x, wp, cmask_ws, pack,
                                                              slad_part, out);
  slad_reduce<<<BATCH, 256, 0, stream>>>(slad_part, out + (size_t)NSAMP * DIM);
}
